// AttentionLayer_54760833024546
// MI455X (gfx1250) — matrix operand order repaired and checked by reading, not compiled, not run on hardware
//
#include <hip/hip_runtime.h>

// Problem constants (match reference)
#define NB 4
#define SL 2048
#define TL 2048
#define DD 1024

typedef __bf16 bf16;
typedef __attribute__((ext_vector_type(8)))  __bf16 v8bf;
typedef __attribute__((ext_vector_type(16))) __bf16 v16bf;
typedef __attribute__((ext_vector_type(8)))  float  v8f;
typedef __attribute__((ext_vector_type(4)))  int    i32x4;

#define AS1 __attribute__((address_space(1)))
#define AS3 __attribute__((address_space(3)))

#if __has_builtin(__builtin_amdgcn_global_load_async_to_lds_b128) && \
    __has_builtin(__builtin_amdgcn_s_wait_asynccnt)
#define USE_ASYNC_LDS 1
#endif

// Dynamic LDS layout (bytes). q rows padded to 1032 elements so consecutive
// rows are offset by 2064B = 129*16B -> ds_load_b128 across 16 rows hits
// distinct bank groups.
#define SQ_STRIDE   1032
#define OFF_SCORES  66048   /* 32*1032*2 */
#define OFF_PROBS   74368   /* + 32*65*4 */
#define OFF_ALPHA   78976   /* + 32*72*2 */
#define OFF_L       79104
#define SMEM_TOTAL  79232

__device__ __forceinline__ v8f wmma_bf16(v16bf a, v16bf b, v8f c) {
  return __builtin_amdgcn_wmma_f32_16x16x32_bf16(
      false, a, false, b, (short)0, c, false, false);
}

__device__ __forceinline__ v16bf frag_a_bf16(const bf16* plo, const bf16* phi) {
  v8bf lo = *(const v8bf*)plo;
  v8bf hi = *(const v8bf*)phi;
  v16bf r;
#pragma unroll
  for (int i = 0; i < 8; ++i) { r[i] = lo[i]; r[i + 8] = hi[i]; }
  return r;
}

__device__ __forceinline__ v16bf frag_a_f32(const float* plo, const float* phi) {
  v16bf r;
#pragma unroll
  for (int i = 0; i < 8; ++i) r[i] = (bf16)plo[i];
#pragma unroll
  for (int i = 0; i < 8; ++i) r[i + 8] = (bf16)phi[i];
  return r;
}

__device__ __forceinline__ v16bf frag_b_f32(const float* p) {
  v16bf r;
#pragma unroll
  for (int i = 0; i < 16; ++i) r[i] = (bf16)p[i];
  return r;
}

// ---------------------------------------------------------------------------
// Kernel 1: P = X @ W^T + bias, bf16 out.  z=0 -> qp, z=1 -> kp, z=2 -> vt
// (value stored TRANSPOSED [N, D, T] so probs@v B-frags are contiguous).
// ---------------------------------------------------------------------------
__global__ __launch_bounds__(256, 1) void proj_gemm(
    const float* __restrict__ q, const float* __restrict__ k,
    const float* __restrict__ v, const float* __restrict__ W,
    const float* __restrict__ bias,
    bf16* __restrict__ qp, bf16* __restrict__ kp, bf16* __restrict__ vt)
{
  const int wave = threadIdx.x >> 5;
  const int lane = threadIdx.x & 31;
  const int hi   = lane >> 4;
  const int lm   = lane & 15;
  const int wm   = wave >> 2;
  const int we   = wave & 3;
  const int row0 = blockIdx.x * 64 + wm * 32;
  const int e0   = blockIdx.y * 256 + we * 64;
  const float* src = (blockIdx.z == 0) ? q : (blockIdx.z == 1) ? k : v;

  v8f acc[2][4];
#pragma unroll
  for (int mi = 0; mi < 2; ++mi)
#pragma unroll
    for (int ni = 0; ni < 4; ++ni) acc[mi][ni] = v8f{};

  const float* arow[2];
  arow[0] = src + (size_t)(row0 + lm) * DD;
  arow[1] = src + (size_t)(row0 + 16 + lm) * DD;
  const float* brow[4];
#pragma unroll
  for (int ni = 0; ni < 4; ++ni)
    brow[ni] = W + (size_t)(e0 + ni * 16 + lm) * DD;

  for (int k0 = 0; k0 < DD; k0 += 32) {
    v16bf af[2];
#pragma unroll
    for (int mi = 0; mi < 2; ++mi)
      af[mi] = frag_a_f32(arow[mi] + k0 + hi * 16, arow[mi] + k0 + hi * 16 + 8);
    v16bf bf_[4];
#pragma unroll
    for (int ni = 0; ni < 4; ++ni)
      bf_[ni] = frag_b_f32(brow[ni] + k0 + hi * 16);
#pragma unroll
    for (int mi = 0; mi < 2; ++mi)
#pragma unroll
      for (int ni = 0; ni < 4; ++ni)
        acc[mi][ni] = wmma_bf16(af[mi], bf_[ni], acc[mi][ni]);
  }

#pragma unroll
  for (int mi = 0; mi < 2; ++mi)
#pragma unroll
    for (int ni = 0; ni < 4; ++ni) {
      const int col = e0 + ni * 16 + lm;
      const float bb = bias[col];
#pragma unroll
      for (int r = 0; r < 8; ++r) {
        const int row = row0 + mi * 16 + hi * 8 + r;
        const float val = acc[mi][ni][r] + bb;
        if (blockIdx.z == 0)      qp[(size_t)row * DD + col] = (bf16)val;
        else if (blockIdx.z == 1) kp[(size_t)row * DD + col] = (bf16)val;
        else {
          const int n = row >> 11;
          const int t = row & (SL - 1);
          vt[((size_t)n * DD + col) * TL + t] = (bf16)val;
        }
      }
    }
}

// ---------------------------------------------------------------------------
// Kernel 2: fused flash attention.
//  - q block (32 x 1024 bf16) staged once in LDS (async-to-LDS if available)
//  - per 64-wide T tile: 1 score WMMA tile/wave -> LDS -> parallel online
//    softmax (8 lanes/row, shfl_xor reductions) -> alpha-rescale + probs@v
// ---------------------------------------------------------------------------
__global__ __launch_bounds__(256, 1) void attn_fused(
    const bf16* __restrict__ qp, const bf16* __restrict__ kp,
    const bf16* __restrict__ vt, float* __restrict__ out)
{
  extern __shared__ __align__(16) char smem[];
  bf16*  sq       = (bf16*)smem;
  float* s_scores = (float*)(smem + OFF_SCORES);   // [32][65]
  bf16*  s_probs  = (bf16*)(smem + OFF_PROBS);     // [32][72]
  float* s_alpha  = (float*)(smem + OFF_ALPHA);    // [32]
  float* s_l      = (float*)(smem + OFF_L);        // [32]

  const int tid  = threadIdx.x;
  const int n    = blockIdx.y;
  const int s0   = blockIdx.x * 32;
  const int wave = tid >> 5;
  const int lane = tid & 31;
  const int hi   = lane >> 4;
  const int lm   = lane & 15;
  const int smi  = wave >> 2;
  const int sni  = wave & 3;
  const int e0   = wave * 128;

  const bf16* qbase = qp + ((size_t)n * SL + s0) * DD;
  const bf16* kbase = kp + (size_t)n * TL * DD;
  const bf16* vbase = vt + (size_t)n * DD * TL;

  // ---- Stage q block into LDS (32 rows x 1024 el = 4096 x 16B chunks) ----
#pragma unroll
  for (int it = 0; it < 16; ++it) {
    const int idx = it * 256 + tid;
    const int row = idx >> 7;
    const int ch  = idx & 127;
    const bf16* g = qbase + (size_t)row * DD + ch * 8;
    bf16*       l = sq + row * SQ_STRIDE + ch * 8;
#ifdef USE_ASYNC_LDS
    __builtin_amdgcn_global_load_async_to_lds_b128(
        (AS1 i32x4*)(void*)g, (AS3 i32x4*)(void*)l, 0, 0);
#else
    *(v8bf*)l = *(const v8bf*)g;
#endif
  }
#ifdef USE_ASYNC_LDS
  __builtin_amdgcn_s_wait_asynccnt(0);
#endif
  __syncthreads();

  v8f acc[2][8];
#pragma unroll
  for (int mi = 0; mi < 2; ++mi)
#pragma unroll
    for (int nj = 0; nj < 8; ++nj) acc[mi][nj] = v8f{};

  float mrun = -1e30f, lrun = 0.0f;
  const float scale = 0.03125f;   // 1/sqrt(1024)

  const bf16* aq = sq + (smi * 16 + lm) * SQ_STRIDE;
  const int   srow = tid >> 3;    // softmax: row 0..31
  const int   ssub = tid & 7;     // 8 lanes per row

  for (int j = 0; j < TL; j += 64) {
    // ---- Phase 1: score tile (16x16 per wave), K over D=1024 ----
    const bf16* kb = kbase + (size_t)(j + sni * 16 + lm) * DD;
    if (j + 64 < TL) {            // prefetch next k tile rows
      const bf16* kn = kbase + (size_t)(j + 64 + sni * 16 + lm) * DD;
      __builtin_prefetch(kn, 0, 1);
      __builtin_prefetch(kn + 512, 0, 1);
    }
    v8f sacc = v8f{};
#pragma unroll 8
    for (int k0 = 0; k0 < DD; k0 += 32) {
      v16bf a = frag_a_bf16(aq + k0 + hi * 16, aq + k0 + hi * 16 + 8);
      v16bf b = *(const v16bf*)(kb + k0 + hi * 16);
      sacc = wmma_bf16(a, b, sacc);
    }
#pragma unroll
    for (int r = 0; r < 8; ++r)
      s_scores[(smi * 16 + hi * 8 + r) * 65 + sni * 16 + lm] = sacc[r];
    __syncthreads();

    // ---- Phase 2: online softmax, all 256 threads (8 per row) ----
    {
      float vc[8];
      float mx = mrun;
#pragma unroll
      for (int i = 0; i < 8; ++i) {
        vc[i] = s_scores[srow * 65 + ssub * 8 + i] * scale;
        mx = fmaxf(mx, vc[i]);
      }
#pragma unroll
      for (int off = 1; off < 8; off <<= 1)
        mx = fmaxf(mx, __shfl_xor(mx, off, 32));
      const float al = __expf(mrun - mx);
      float ls = 0.0f;
#pragma unroll
      for (int i = 0; i < 8; ++i) {
        const float p = __expf(vc[i] - mx);
        ls += p;
        s_probs[srow * 72 + ssub * 8 + i] = (bf16)p;
      }
#pragma unroll
      for (int off = 1; off < 8; off <<= 1)
        ls += __shfl_xor(ls, off, 32);
      lrun = lrun * al + ls;
      mrun = mx;
      if (ssub == 0) { s_alpha[srow] = al; s_l[srow] = lrun; }
    }
    __syncthreads();

    // ---- Phase 3: rescale accumulator, acc += probs @ v_slice ----
#pragma unroll
    for (int mi = 0; mi < 2; ++mi) {
      float alr[8];
#pragma unroll
      for (int r = 0; r < 8; ++r) alr[r] = s_alpha[mi * 16 + hi * 8 + r];
#pragma unroll
      for (int nj = 0; nj < 8; ++nj)
#pragma unroll
        for (int r = 0; r < 8; ++r) acc[mi][nj][r] *= alr[r];
    }

#pragma unroll
    for (int kc = 0; kc < 64; kc += 32) {
      v16bf pa[2];
#pragma unroll
      for (int mi = 0; mi < 2; ++mi)
        pa[mi] = frag_a_bf16(&s_probs[(mi * 16 + lm) * 72 + kc + hi * 16],
                             &s_probs[(mi * 16 + lm) * 72 + kc + hi * 16 + 8]);
#pragma unroll
      for (int nj = 0; nj < 8; ++nj) {
        v16bf bv = *(const v16bf*)(vbase + (size_t)(e0 + nj * 16 + lm) * TL
                                   + j + kc + hi * 16);
        acc[0][nj] = wmma_bf16(pa[0], bv, acc[0][nj]);
        acc[1][nj] = wmma_bf16(pa[1], bv, acc[1][nj]);
      }
    }
    __syncthreads();
  }

  // ---- Epilogue: divide by running sum, store f32 ----
  float linv[2][8];
#pragma unroll
  for (int mi = 0; mi < 2; ++mi)
#pragma unroll
    for (int r = 0; r < 8; ++r)
      linv[mi][r] = 1.0f / s_l[mi * 16 + hi * 8 + r];

#pragma unroll
  for (int mi = 0; mi < 2; ++mi)
#pragma unroll
    for (int nj = 0; nj < 8; ++nj) {
      const int col = e0 + nj * 16 + lm;
#pragma unroll
      for (int r = 0; r < 8; ++r) {
        const int row = s0 + mi * 16 + hi * 8 + r;
        out[((size_t)n * SL + row) * DD + col] = acc[mi][nj][r] * linv[mi][r];
      }
    }
}

// ---------------------------------------------------------------------------
extern "C" void kernel_launch(void* const* d_in, const int* in_sizes, int n_in,
                              void* d_out, int out_size, void* d_ws, size_t ws_size,
                              hipStream_t stream) {
  const float* q    = (const float*)d_in[0];
  const float* k    = (const float*)d_in[1];
  const float* v    = (const float*)d_in[2];
  const float* W    = (const float*)d_in[3];
  const float* bias = (const float*)d_in[4];
  float* out = (float*)d_out;

  bf16* qp = (bf16*)d_ws;
  bf16* kp = qp + (size_t)NB * SL * DD;
  bf16* vt = kp + (size_t)NB * TL * DD;

  proj_gemm<<<dim3(128, 4, 3), 256, 0, stream>>>(q, k, v, W, bias, qp, kp, vt);
  attn_fused<<<dim3(SL / 32, NB), 256, SMEM_TOTAL, stream>>>(qp, kp, vt, out);
}
